// FlashAttention_52201032515820
// MI455X (gfx1250) — matrix operand order repaired and checked by reading, not compiled, not run on hardware
//
#include <hip/hip_runtime.h>
#include <stdint.h>

// ---------------------------------------------------------------------------
// Problem constants (from reference): B=4, S=4096, D=1024, H=16, HEAD_DIM=192
//   GEMM: [16384 x 1024] * [1024 x 3072] -> qkv
//   Attention per position: 16x16 over heads, contraction 64, softmax, P*V
// ---------------------------------------------------------------------------

typedef __attribute__((ext_vector_type(16))) __bf16   v16bf;
typedef __attribute__((ext_vector_type(8)))  float    v8f;
typedef __attribute__((ext_vector_type(4)))  uint32_t u32x4;
typedef __attribute__((ext_vector_type(4)))  float    f32x4;
typedef __attribute__((ext_vector_type(2)))  float    f32x2;

union FragB {
    v16bf    v;
    uint32_t w[8];
    u32x4    q[2];
};

__device__ __forceinline__ uint32_t bf16_bits(float f) {
    uint32_t u = __builtin_bit_cast(uint32_t, f);
    return (u + 0x7FFFu + ((u >> 16) & 1u)) >> 16;   // round-to-nearest-even
}
__device__ __forceinline__ uint32_t pack_bf16(float a, float b) {
    return (bf16_bits(a) & 0xFFFFu) | (bf16_bits(b) << 16);
}
__device__ __forceinline__ v8f wmma_bf16(FragB a, FragB b, v8f c) {
    return __builtin_amdgcn_wmma_f32_16x16x32_bf16(
        /*neg_a=*/false, a.v, /*neg_b=*/false, b.v,
        /*c_mod=*/(short)0, c, /*reuse_a=*/false, /*reuse_b=*/false);
}

// ---------------------------------------------------------------------------
// Kernel 1: pack W [1024][3072] f32 -> bf16 in WMMA B-fragment order.
// Tile t = ntile*32 + ktile covers K rows [ktile*32,+32), N cols [ntile*16,+16).
// B-frag layout (16-bit, 32x16): VGPR v<4 holds K pair (khalf*8 + 2v, +1),
// v>=4 holds (16 + khalf*8 + 2(v-4), +1); lane's column n = lane&15.
// Packed tile = 512 bf16 (1 KB): dword index = t*256 + lane*8 + v.
// ---------------------------------------------------------------------------
__global__ __launch_bounds__(256) void wpack_kernel(const float* __restrict__ W,
                                                    uint32_t* __restrict__ Wp) {
    int gt    = blockIdx.x * 256 + threadIdx.x;
    int t     = gt >> 5;          // tile id: 0..6143
    int lane  = gt & 31;
    int ntile = t >> 5;           // 0..191
    int ktile = t & 31;           // 0..31
    int nhalf = lane >> 4;
    int col   = ntile * 16 + (lane & 15);
    uint32_t o[8];
#pragma unroll
    for (int v = 0; v < 8; ++v) {
        int k   = ktile * 32 + nhalf * 8 + (v & 3) * 2 + (v >> 2) * 16;
        float a = W[(size_t)k * 3072 + col];
        float b = W[(size_t)(k + 1) * 3072 + col];
        o[v]    = pack_bf16(a, b);
    }
    u32x4 lo4, hi4;
    lo4.x = o[0]; lo4.y = o[1]; lo4.z = o[2]; lo4.w = o[3];
    hi4.x = o[4]; hi4.y = o[5]; hi4.z = o[6]; hi4.w = o[7];
    u32x4* dst = (u32x4*)(Wp + (size_t)t * 256 + lane * 8);
    dst[0] = lo4;
    dst[1] = hi4;
}

// ---------------------------------------------------------------------------
// Kernel 2: fused qkv-GEMM + per-position head attention.
// One block = 256 threads (8 waves) handles 32 sequence positions.
// LDS: xs bf16 [32][1024] (64KB) | qkv bf16 [32][3072] (192KB) | ps f32 8x256 (8KB)
// ---------------------------------------------------------------------------
__global__ __launch_bounds__(256, 1) void qkv_attn_kernel(
    const float* __restrict__ x, const uint32_t* __restrict__ Wp,
    const float* __restrict__ bias, float* __restrict__ out) {
    extern __shared__ __align__(16) char smem[];
    uint32_t* xs32  = (uint32_t*)smem;             // 32 x 512 dwords (bf16 pairs)
    uint16_t* qkv16 = (uint16_t*)(smem + 65536);   // 32 x 3072 bf16
    uint32_t* qkv32 = (uint32_t*)(smem + 65536);
    float*    ps    = (float*)(smem + 262144);     // per-wave 16x16 f32 scratch

    const int  tid  = threadIdx.x;
    const int  wave = tid >> 5;
    const int  lane = tid & 31;
    const int  lm   = lane & 15;   // column / row-within-half
    const int  lh   = lane >> 4;   // half-wave selector
    const long pos0 = (long)blockIdx.x * 32;

    // ---------- Phase A: x tile [32][1024] f32 -> bf16 LDS ----------
    const f32x4* xg = (const f32x4*)(x + pos0 * 1024);
#pragma unroll
    for (int j = 0; j < 32; ++j) {
        int   i = j * 256 + tid;           // float4 index, fully coalesced
        f32x4 f = xg[i];
        xs32[i * 2]     = pack_bf16(f.x, f.y);
        xs32[i * 2 + 1] = pack_bf16(f.z, f.w);
    }
    __syncthreads();

    // ---------- Phase B: GEMM, each wave owns 24 N-tiles ----------
    // A-frag (16x32 bf16): lane row m = lm; v0-3: K = 2v + lh*8, v4-7: K = 16+2(v-4)+lh*8
    // -> dwords [m*512 + kt*16 + lh*4 .. +3] and [+8 .. +11] : two b128 LDS loads.
    const u32x4* WpQ = (const u32x4*)Wp;
    const int    nt0 = wave * 24;
#pragma unroll 1
    for (int g = 0; g < 24; g += 2) {
        v8f    acc00 = {}, acc01 = {}, acc10 = {}, acc11 = {};
        size_t bA = (size_t)(nt0 + g) * 2048 + lane * 2;   // u32x4 index
        size_t bB = bA + 2048;
        const int rb = lm * 512;
#pragma unroll 4
        for (int kt = 0; kt < 32; ++kt) {
            FragB a0, a1, b0, b1;
            const uint32_t* xr = xs32 + rb + kt * 16 + lh * 4;
            a0.q[0] = *(const u32x4*)xr;
            a0.q[1] = *(const u32x4*)(xr + 8);
            a1.q[0] = *(const u32x4*)(xr + 8192);      // M-subtile 1 (+16 rows)
            a1.q[1] = *(const u32x4*)(xr + 8200);
            b0.q[0] = WpQ[bA + (size_t)kt * 64];
            b0.q[1] = WpQ[bA + (size_t)kt * 64 + 1];
            b1.q[0] = WpQ[bB + (size_t)kt * 64];
            b1.q[1] = WpQ[bB + (size_t)kt * 64 + 1];
            if (kt + 4 < 32) {
                __builtin_prefetch((const void*)(WpQ + bA + (size_t)(kt + 4) * 64), 0, 0);
                __builtin_prefetch((const void*)(WpQ + bB + (size_t)(kt + 4) * 64), 0, 0);
            }
            acc00 = wmma_bf16(a0, b0, acc00);
            acc10 = wmma_bf16(a1, b0, acc10);
            acc01 = wmma_bf16(a0, b1, acc01);
            acc11 = wmma_bf16(a1, b1, acc11);
        }
        // C layout: VGPR e, lane -> row m = e + 8*lh, col n = lm. Add bias, store bf16.
#pragma unroll
        for (int j = 0; j < 2; ++j) {
            int   nn = (nt0 + g + j) * 16 + lm;
            float bv = bias[nn];
            v8f   aL = j ? acc01 : acc00;
            v8f   aH = j ? acc11 : acc10;
#pragma unroll
            for (int e = 0; e < 8; ++e) {
                int r0 = e + 8 * lh;
                qkv16[(size_t)r0 * 3072 + nn]        = (uint16_t)bf16_bits(aL[e] + bv);
                qkv16[(size_t)(r0 + 16) * 3072 + nn] = (uint16_t)bf16_bits(aH[e] + bv);
            }
        }
    }
    __syncthreads();

    // ---------- Phase C: attention over heads, 4 positions per wave ----------
    const float SCALE = 0.07216878364870322f;   // 1/sqrt(192) per reference
    float*      myps  = ps + wave * 256;
#pragma unroll 1
    for (int i = 0; i < 4; ++i) {
        const int       p    = wave * 4 + i;
        const uint32_t* qr32 = qkv32 + (size_t)p * 1536;
        const uint16_t* qr16 = qkv16 + (size_t)p * 3072;

        // scores[h][g] = sum_f q[h][f]*k[g][f], K=64 -> two 16x16x32 WMMAs
        v8f sc = {};
#pragma unroll
        for (int kt2 = 0; kt2 < 2; ++kt2) {
            FragB qa, kb;
            const uint32_t* qb = qr32 + lm * 96 + kt2 * 16 + lh * 4;  // q: cols 0..63
            qa.q[0] = *(const u32x4*)qb;
            qa.q[1] = *(const u32x4*)(qb + 8);
            const uint32_t* kp = qr32 + lm * 96 + 32 + kt2 * 16 + lh * 4;  // k: cols 64..127
            kb.q[0] = *(const u32x4*)kp;
            kb.q[1] = *(const u32x4*)(kp + 8);
            sc = wmma_bf16(qa, kb, sc);
        }

        // softmax over g: each row lives in one 16-lane half -> xor-butterfly
        float pr[8];
#pragma unroll
        for (int e = 0; e < 8; ++e) {
            float t  = sc[e] * SCALE;
            float mx = t;
            mx = fmaxf(mx, __shfl_xor(mx, 1, 32));
            mx = fmaxf(mx, __shfl_xor(mx, 2, 32));
            mx = fmaxf(mx, __shfl_xor(mx, 4, 32));
            mx = fmaxf(mx, __shfl_xor(mx, 8, 32));
            float ex = __expf(t - mx);
            float sm = ex;
            sm += __shfl_xor(sm, 1, 32);
            sm += __shfl_xor(sm, 2, 32);
            sm += __shfl_xor(sm, 4, 32);
            sm += __shfl_xor(sm, 8, 32);
            pr[e] = ex / sm;
        }

        // transpose P (C layout -> A layout) through per-wave LDS scratch
#pragma unroll
        for (int e = 0; e < 8; ++e) myps[(e + 8 * lh) * 16 + lm] = pr[e];
        asm volatile("s_wait_dscnt 0" ::: "memory");   // same-wave DS ordering fence
        __builtin_amdgcn_wave_barrier();

        FragB pa;   // A-frag 16x32, K=g (valid 0..15, upper half zero-padded)
#pragma unroll
        for (int v = 0; v < 4; ++v) {
            f32x2 f2   = *(const f32x2*)(myps + lm * 16 + v * 2 + lh * 8);
            pa.w[v]     = pack_bf16(f2.x, f2.y);
            pa.w[v + 4] = 0u;
        }

        // out[h][f] = sum_g P[h][g] * V[g][f]; V rows g live at cols 128..191
        float* optr = out + (pos0 + p) * 1024;   // 16 heads * 64
#pragma unroll
        for (int ft = 0; ft < 4; ++ft) {
            FragB vb;   // B-frag: v0-3 hold K=8*lh+2v,+1 (valid 0..15), v4-7 zero
#pragma unroll
            for (int v = 0; v < 8; ++v) {
                int      gg = (2 * v + 8 * lh) & 15;
                uint32_t lo = qr16[(size_t)gg * 192 + 128 + ft * 16 + lm];
                uint32_t hi = qr16[(size_t)(gg + 1) * 192 + 128 + ft * 16 + lm];
                vb.w[v]     = (v >= 4) ? 0u : (lo | (hi << 16));
            }
            v8f o = {};
            o     = wmma_bf16(pa, vb, o);
#pragma unroll
            for (int e = 0; e < 8; ++e)
                optr[(size_t)(e + 8 * lh) * 64 + ft * 16 + lm] = o[e];
        }
    }
}

// ---------------------------------------------------------------------------
extern "C" void kernel_launch(void* const* d_in, const int* in_sizes, int n_in,
                              void* d_out, int out_size, void* d_ws, size_t ws_size,
                              hipStream_t stream) {
    (void)in_sizes; (void)n_in; (void)out_size; (void)ws_size;
    const float* x    = (const float*)d_in[0];   // [4,4096,1024] f32
    const float* W    = (const float*)d_in[1];   // [1024,16,192] f32
    const float* bias = (const float*)d_in[2];   // [16,192] f32
    float*       out  = (float*)d_out;           // [4,4096,16,64] f32
    uint32_t*    Wp   = (uint32_t*)d_ws;         // 6 MiB packed bf16 weights

    // 6144 tiles * 1 wave each = 768 blocks of 256 threads
    wpack_kernel<<<768, 256, 0, stream>>>(W, Wp);

    (void)hipFuncSetAttribute((const void*)qkv_attn_kernel,
                              hipFuncAttributeMaxDynamicSharedMemorySize, 270336);
    // 16384 positions / 32 per block = 512 blocks; 270336 B dynamic LDS
    qkv_attn_kernel<<<512, 256, 270336, stream>>>(x, Wp, bias, out);
}
